// TriangleAttentionStartingNode_30270929502439
// MI455X (gfx1250) — matrix operand order repaired and checked by reading, not compiled, not run on hardware
//
#include <hip/hip_runtime.h>
#include <hip/hip_bf16.h>
#include <math.h>

// ---------------------------------------------------------------------------
// Triangle attention (starting node) for MI455X / gfx1250, wave32 + WMMA f16.
// B=1, N=320, C=128, H=4, D=32.  Memory-bound => fp16 intermediates,
// v_wmma_f32_16x16x32_f16 for all matmuls, LN fused into projection GEMM,
// async-to-LDS staging for K via global_load_async_to_lds_b128.
// ---------------------------------------------------------------------------

typedef __attribute__((ext_vector_type(16))) _Float16 v16h;
typedef __attribute__((ext_vector_type(8)))  _Float16 v8h;
typedef __attribute__((ext_vector_type(8)))  float    v8f;

#if __has_builtin(__builtin_amdgcn_global_load_async_to_lds_b128)
#define HAVE_ASYNC_LDS 1
typedef int v4i __attribute__((vector_size(16)));
typedef __attribute__((address_space(1))) v4i* gp_v4i;   // pointer to int4 in global AS
typedef __attribute__((address_space(3))) v4i* lp_v4i;   // pointer to int4 in LDS AS
#endif

#define WMMA(a, b, c) \
  __builtin_amdgcn_wmma_f32_16x16x32_f16(false, (a), false, (b), (short)0, (c), false, false)

#define NSEQ   320
#define NROWS  (NSEQ * NSEQ)          // 102400 flattened (i,j) rows
#define CCH    128
#define NHEAD  4
#define DHEAD  32
#define QKVG_LD 512                   // q[0:128) k[128:256) v[256:384) g[384:512)
#define NTILE_IN 33                   // 528 cols: q,k,v,g (512) + wb (4) + pad
#define NTILE_OUT 8

static __device__ __forceinline__ v16h cat16(v8h lo, v8h hi) {
  v16h r;
  #pragma unroll
  for (int j = 0; j < 8; ++j) { r[j] = lo[j]; r[j + 8] = hi[j]; }
  return r;
}

// fragment-ordered weight buffer: ((ntile*4 + kt)*32 + lane)*16 + j
//   value = W[k = kt*32 + (lane>>4)*8 + j + (j>=8 ? 8 : 0)][n = ntile*16 + (lane&15)]
__global__ __launch_bounds__(256)
void prep_weights(const float* __restrict__ wq, const float* __restrict__ wk,
                  const float* __restrict__ wv, const float* __restrict__ wb,
                  const float* __restrict__ wg, const float* __restrict__ wo,
                  _Float16* __restrict__ wfrag_in, _Float16* __restrict__ wfrag_out) {
  const int IN_TOT  = NTILE_IN  * 4 * 32 * 16;  // 67584
  const int OUT_TOT = NTILE_OUT * 4 * 32 * 16;  // 16384
  int idx = blockIdx.x * blockDim.x + threadIdx.x;
  if (idx < IN_TOT) {
    int j = idx & 15, lane = (idx >> 4) & 31, kt = (idx >> 9) & 3, nt = idx >> 11;
    int k = kt * 32 + (lane >> 4) * 8 + j + (j >= 8 ? 8 : 0);
    int n = nt * 16 + (lane & 15);
    float val = 0.f;
    if      (n < 128) val = wq[k * 128 +  n       ] * 0.17677669529663687f; // 1/sqrt(32)
    else if (n < 256) val = wk[k * 128 + (n - 128)];
    else if (n < 384) val = wv[k * 128 + (n - 256)];
    else if (n < 512) val = wg[k * 128 + (n - 384)];
    else if (n < 516) val = wb[k * 4   + (n - 512)];
    wfrag_in[idx] = (_Float16)val;
  } else if (idx < IN_TOT + OUT_TOT) {
    int t = idx - IN_TOT;
    int j = t & 15, lane = (t >> 4) & 31, kt = (t >> 9) & 3, nt = t >> 11;
    int k = kt * 32 + (lane >> 4) * 8 + j + (j >= 8 ? 8 : 0);
    int n = nt * 16 + (lane & 15);
    wfrag_out[t] = (_Float16)wo[k * 128 + n];
  }
}

// One wave per 16-row tile: fused LayerNorm + [q|k|v|g|b] projection GEMM.
__global__ __launch_bounds__(32)
void ln_proj(const float* __restrict__ x2d, const float* __restrict__ ln_scale,
             const float* __restrict__ ln_bias, const float* __restrict__ bg,
             const _Float16* __restrict__ wfrag_in,
             _Float16* __restrict__ qkvg, float* __restrict__ bias_tj) {
  __shared__ _Float16 xn[16][CCH];
  const int lane = threadIdx.x;
  const long rowbase = (long)blockIdx.x * 16;

  // ---- LayerNorm: 16 rows, whole wave per row -----------------------------
  for (int r = 0; r < 16; ++r) {
    const float* xr = x2d + (rowbase + r) * CCH;
    float v0 = xr[lane], v1 = xr[lane + 32], v2 = xr[lane + 64], v3 = xr[lane + 96];
    float s  = v0 + v1 + v2 + v3;
    float s2 = v0 * v0 + v1 * v1 + v2 * v2 + v3 * v3;
    #pragma unroll
    for (int off = 16; off >= 1; off >>= 1) {
      s  += __shfl_xor(s,  off, 32);
      s2 += __shfl_xor(s2, off, 32);
    }
    float mu   = s * (1.f / 128.f);
    float var  = s2 * (1.f / 128.f) - mu * mu;
    float rstd = rsqrtf(var + 1e-5f);
    xn[r][lane     ] = (_Float16)((v0 - mu) * rstd * ln_scale[lane     ] + ln_bias[lane     ]);
    xn[r][lane + 32] = (_Float16)((v1 - mu) * rstd * ln_scale[lane + 32] + ln_bias[lane + 32]);
    xn[r][lane + 64] = (_Float16)((v2 - mu) * rstd * ln_scale[lane + 64] + ln_bias[lane + 64]);
    xn[r][lane + 96] = (_Float16)((v3 - mu) * rstd * ln_scale[lane + 96] + ln_bias[lane + 96]);
  }
  __syncthreads();

  // ---- A fragments (16x32 f16 per K-step, ISA 7.12.2 layout) --------------
  const int mloc = lane & 15, khalf = lane >> 4, kb = khalf * 8;
  v16h a[4];
  #pragma unroll
  for (int kt = 0; kt < 4; ++kt) {
    const _Float16* xr = &xn[mloc][kt * 32 + kb];
    a[kt] = cat16(*(const v8h*)xr, *(const v8h*)(xr + 16));
  }

  // ---- 33 output n-tiles x 4 K-steps of WMMA ------------------------------
  for (int nt = 0; nt < NTILE_IN; ++nt) {
    v16h b[4];
    #pragma unroll
    for (int kt = 0; kt < 4; ++kt)
      b[kt] = *(const v16h*)(wfrag_in + ((nt * 4 + kt) * 32 + lane) * 16);
    v8f acc = {};
    #pragma unroll
    for (int kt = 0; kt < 4; ++kt) acc = WMMA(a[kt], b[kt], acc);

    int n = nt * 16 + mloc;
    if (nt < 24) {                       // q / k / v -> f16
      #pragma unroll
      for (int r = 0; r < 8; ++r) {
        long grow = rowbase + r + 8 * khalf;
        qkvg[grow * QKVG_LD + n] = (_Float16)acc[r];
      }
    } else if (nt < 32) {                // g -> sigmoid(acc + bg), f16
      float bgv = bg[n - 384];
      #pragma unroll
      for (int r = 0; r < 8; ++r) {
        long grow = rowbase + r + 8 * khalf;
        float z = acc[r] + bgv;
        qkvg[grow * QKVG_LD + n] = (_Float16)__builtin_amdgcn_rcpf(1.f + __expf(-z));
      }
    } else {                             // pair bias, transposed: [h][j*320+i]
      if (mloc < 4) {
        #pragma unroll
        for (int r = 0; r < 8; ++r) {
          long grow = rowbase + r + 8 * khalf;
          int ii = (int)(grow / NSEQ);
          int jj = (int)(grow - (long)ii * NSEQ);
          bias_tj[(long)mloc * NROWS + (long)jj * NSEQ + ii] = acc[r];
        }
      }
    }
  }
}

// One block per (h, m): full 320x320 attention, WMMA both GEMMs.
// Dynamic LDS: K[320][32]f16 | Vt[32][328]f16 (transposed) | S[4][16][324]f32
__global__ __launch_bounds__(128)
void attention(const _Float16* __restrict__ qkvg, const float* __restrict__ bias_tj,
               _Float16* __restrict__ o_buf) {
  extern __shared__ char smem[];
  _Float16* Klds = (_Float16*)smem;                 // 320*32*2  = 20480 B
  _Float16* Vt   = (_Float16*)(smem + 20480);       // 32*328*2  = 20992 B
  float*    S    = (float*)(smem + 41472);          // 4*16*324*4= 82944 B
  const int SLD = 324;                              // padded S row stride (dwords)
  const int VLD = 328;                              // padded Vt row stride (f16)

  const int h = blockIdx.x;
  const int m = blockIdx.y;
  const int tid = threadIdx.x;
  const int lane = tid & 31, wid = tid >> 5;
  const int mloc = lane & 15, khalf = lane >> 4, kb = khalf * 8;
  float* Sw = S + wid * 16 * SLD;

  // ---- stage K (async-to-LDS) and V (transposed) --------------------------
  for (int idx = tid; idx < NSEQ * 4; idx += 128) {
    int row = idx >> 2, part = idx & 3;
    const _Float16* src = qkvg + ((long)(m * NSEQ + row)) * QKVG_LD + h * DHEAD + part * 8;
#if HAVE_ASYNC_LDS
    __builtin_amdgcn_global_load_async_to_lds_b128(
        (gp_v4i)(src + 128), (lp_v4i)&Klds[row * DHEAD + part * 8], 0, 0);
#else
    *(uint4*)&Klds[row * DHEAD + part * 8] = *(const uint4*)(src + 128);
#endif
    v8h vv = *(const v8h*)(src + 256);
    #pragma unroll
    for (int e = 0; e < 8; ++e) Vt[(part * 8 + e) * VLD + row] = vv[e];
  }
#if HAVE_ASYNC_LDS
#if __has_builtin(__builtin_amdgcn_s_wait_asynccnt)
  __builtin_amdgcn_s_wait_asynccnt(0);
#else
  asm volatile("s_wait_asynccnt 0x0" ::: "memory");
#endif
#endif
  __syncthreads();

  for (int it = wid; it < 20; it += 4) {            // 5 i-tiles per wave
    // Q fragment: A 16x32, K-dim = D = 32 (single WMMA step)
    const _Float16* qrow = qkvg + ((long)(m * NSEQ + it * 16 + mloc)) * QKVG_LD + h * DHEAD;
    v16h aq = cat16(*(const v8h*)(qrow + kb), *(const v8h*)(qrow + kb + 16));

    // ---- S = Q K^T + bias -------------------------------------------------
    for (int jt = 0; jt < 20; ++jt) {
      const _Float16* krow = Klds + (jt * 16 + mloc) * DHEAD + kb;
      v16h bk = cat16(*(const v8h*)krow, *(const v8h*)(krow + 16));
      v8f s = {};
      s = WMMA(aq, bk, s);
      int jcol = jt * 16 + mloc;
      const float4* bj = (const float4*)(bias_tj + (long)h * NROWS +
                                         (long)jcol * NSEQ + it * 16 + 8 * khalf);
      float4 bA = bj[0], bB = bj[1];
      float bb[8] = {bA.x, bA.y, bA.z, bA.w, bB.x, bB.y, bB.z, bB.w};
      #pragma unroll
      for (int r = 0; r < 8; ++r)
        Sw[(r + 8 * khalf) * SLD + jcol] = s[r] + bb[r];
    }

    // ---- softmax over j (row = mloc, lane-halves own 160 cols each) -------
    float4* srow4 = (float4*)(Sw + mloc * SLD + khalf * 160);
    float mx = -3.4e38f;
    for (int c = 0; c < 40; ++c) {
      float4 vv = srow4[c];
      mx = fmaxf(mx, fmaxf(fmaxf(vv.x, vv.y), fmaxf(vv.z, vv.w)));
    }
    mx = fmaxf(mx, __shfl_xor(mx, 16, 32));
    float sum = 0.f;
    for (int c = 0; c < 40; ++c) {
      float4 vv = srow4[c];
      vv.x = __expf(vv.x - mx); vv.y = __expf(vv.y - mx);
      vv.z = __expf(vv.z - mx); vv.w = __expf(vv.w - mx);
      srow4[c] = vv;
      sum += vv.x + vv.y + vv.z + vv.w;
    }
    sum += __shfl_xor(sum, 16, 32);
    float inv = __builtin_amdgcn_rcpf(sum);

    // ---- O = P V (1/sum folded into f16 conversion), gate, store ----------
    #pragma unroll
    for (int dt = 0; dt < 2; ++dt) {
      v8f oacc = {};
      for (int kt = 0; kt < 10; ++kt) {
        const float* prow = Sw + mloc * SLD + kt * 32 + kb;
        float4 p0 = *(const float4*)prow;
        float4 p1 = *(const float4*)(prow + 4);
        float4 p2 = *(const float4*)(prow + 16);
        float4 p3 = *(const float4*)(prow + 20);
        float pv[16] = {p0.x, p0.y, p0.z, p0.w, p1.x, p1.y, p1.z, p1.w,
                        p2.x, p2.y, p2.z, p2.w, p3.x, p3.y, p3.z, p3.w};
        v16h ap;
        #pragma unroll
        for (int j = 0; j < 16; ++j) ap[j] = (_Float16)(pv[j] * inv);
        const _Float16* vrow = Vt + (dt * 16 + mloc) * VLD + kt * 32 + kb;
        v16h bv = cat16(*(const v8h*)vrow, *(const v8h*)(vrow + 16));
        oacc = WMMA(ap, bv, oacc);
      }
      #pragma unroll
      for (int r = 0; r < 8; ++r) {
        int irow = it * 16 + r + 8 * khalf;
        long grow = (long)m * NSEQ + irow;
        float gv = (float)qkvg[grow * QKVG_LD + 384 + h * DHEAD + dt * 16 + mloc];
        o_buf[grow * CCH + h * DHEAD + dt * 16 + mloc] = (_Float16)(oacc[r] * gv);
      }
    }
  }
}

// out = o_f16 @ wo + bo   (f32 output)
__global__ __launch_bounds__(32)
void out_proj(const _Float16* __restrict__ o_buf, const _Float16* __restrict__ wfrag_out,
              const float* __restrict__ bo, float* __restrict__ out) {
  const int lane = threadIdx.x;
  const long rowbase = (long)blockIdx.x * 16;
  const int mloc = lane & 15, khalf = lane >> 4, kb = khalf * 8;
  const _Float16* orow = o_buf + (rowbase + mloc) * CCH;
  v16h a[4];
  #pragma unroll
  for (int kt = 0; kt < 4; ++kt) {
    const _Float16* xr = orow + kt * 32 + kb;
    a[kt] = cat16(*(const v8h*)xr, *(const v8h*)(xr + 16));
  }
  for (int nt = 0; nt < NTILE_OUT; ++nt) {
    v16h b[4];
    #pragma unroll
    for (int kt = 0; kt < 4; ++kt)
      b[kt] = *(const v16h*)(wfrag_out + ((nt * 4 + kt) * 32 + lane) * 16);
    v8f acc = {};
    #pragma unroll
    for (int kt = 0; kt < 4; ++kt) acc = WMMA(a[kt], b[kt], acc);
    int n = nt * 16 + mloc;
    float bias = bo[n];
    #pragma unroll
    for (int r = 0; r < 8; ++r) {
      long grow = rowbase + r + 8 * khalf;
      out[grow * CCH + n] = acc[r] + bias;
    }
  }
}

extern "C" void kernel_launch(void* const* d_in, const int* in_sizes, int n_in,
                              void* d_out, int out_size, void* d_ws, size_t ws_size,
                              hipStream_t stream) {
  const float* x2d      = (const float*)d_in[0];
  const float* ln_scale = (const float*)d_in[1];
  const float* ln_bias  = (const float*)d_in[2];
  const float* wq       = (const float*)d_in[3];
  const float* wk       = (const float*)d_in[4];
  const float* wv       = (const float*)d_in[5];
  const float* wb       = (const float*)d_in[6];
  const float* wg       = (const float*)d_in[7];
  const float* bg       = (const float*)d_in[8];
  const float* wo       = (const float*)d_in[9];
  const float* bo       = (const float*)d_in[10];
  float* out = (float*)d_out;

  char* ws = (char*)d_ws;
  _Float16* wfrag_in  = (_Float16*)(ws);                 //   135,168 B
  _Float16* wfrag_out = (_Float16*)(ws + 135168);        //    32,768 B
  float*    bias_tj   = (float*)   (ws + 167936);        // 1,638,400 B
  _Float16* qkvg      = (_Float16*)(ws + 1806336);       // 104,857,600 B
  _Float16* o_buf     = (_Float16*)(ws + 106663936);     // 26,214,400 B (end 132,878,336)

  prep_weights<<<dim3(328), dim3(256), 0, stream>>>(wq, wk, wv, wb, wg, wo,
                                                    wfrag_in, wfrag_out);
  ln_proj<<<dim3(NROWS / 16), dim3(32), 0, stream>>>(x2d, ln_scale, ln_bias, bg,
                                                     wfrag_in, qkvg, bias_tj);
  attention<<<dim3(NHEAD, NSEQ), dim3(128), 124416, stream>>>(qkvg, bias_tj, o_buf);
  out_proj<<<dim3(NROWS / 16), dim3(32), 0, stream>>>(o_buf, wfrag_out, bo, out);
}
